// Binarized_MLP_30391188587233
// MI455X (gfx1250) — matrix operand order repaired and checked by reading, not compiled, not run on hardware
//
#include <hip/hip_runtime.h>
#include <hip/hip_bf16.h>
#include <cstdint>
#include <math.h>

// ---------------------------------------------------------------------------
// Binarized MLP forward for MI455X (gfx1250), wave32 + WMMA f16 path.
//   h1 = relu(bn(x  @ sign(w1)^T));  h2 = relu(bn(h1 @ sign(w2)^T));
//   h3 = relu(bn(h2 @ sign(w3)^T));  out = sigmoid(h3 @ sign(w4)^T)
// B = 16384, dims 784 -> 1024 -> 1024 -> 1024 -> 10.
//
// GEMM: LDS double-buffered W tiles, K=64 per stage (16 WMMAs per barrier),
// B-fragments bulk-preloaded with a sched_barrier so DScnt waits graduate
// instead of stalling per WMMA, register double-buffered A-frags, BN column
// stats fused into the epilogue via ds_add_f32 + global atomics.
// ---------------------------------------------------------------------------

typedef _Float16 half_t;
typedef __attribute__((ext_vector_type(8)))  _Float16 v8h;
typedef __attribute__((ext_vector_type(16))) _Float16 v16h;
typedef __attribute__((ext_vector_type(8)))  float    v8f;

#define BATCH   16384
#define HID     1024
#define K1_IN   784
#define K1_PAD  832     // next multiple of 64
#define BN_EPS  1e-5f

#if __has_builtin(__builtin_amdgcn_sched_barrier)
#define SCHED_FENCE() __builtin_amdgcn_sched_barrier(0)
#else
#define SCHED_FENCE() asm volatile("" ::: "memory")
#endif

static __device__ __forceinline__ v16h cat16(v8h lo, v8h hi) {
    return __builtin_shufflevector(lo, hi, 0, 1, 2, 3, 4, 5, 6, 7,
                                           8, 9, 10, 11, 12, 13, 14, 15);
}

// --------------------------- small prep kernels ----------------------------

__global__ void binarize_pad_kernel(const float* __restrict__ w, half_t* __restrict__ out,
                                    int rowsIn, int rowsOut, int kIn, int kOut) {
    size_t n = (size_t)rowsOut * (size_t)kOut;
    for (size_t i = (size_t)blockIdx.x * blockDim.x + threadIdx.x; i < n;
         i += (size_t)gridDim.x * blockDim.x) {
        int r = (int)(i / kOut);
        int c = (int)(i % kOut);
        half_t v = (half_t)0.f;
        if (r < rowsIn && c < kIn)
            v = (w[(size_t)r * kIn + c] >= 0.f) ? (half_t)1.f : (half_t)(-1.f);
        out[i] = v;
    }
}

__global__ void cast_pad_kernel(const float* __restrict__ x, half_t* __restrict__ out,
                                int rows, int kIn, int kOut) {
    size_t n = (size_t)rows * (size_t)kOut;
    for (size_t i = (size_t)blockIdx.x * blockDim.x + threadIdx.x; i < n;
         i += (size_t)gridDim.x * blockDim.x) {
        int r = (int)(i / kOut);
        int c = (int)(i % kOut);
        out[i] = (c < kIn) ? (half_t)x[(size_t)r * kIn + c] : (half_t)0.f;
    }
}

__global__ void zero_kernel(float* __restrict__ p, int n) {
    int i = blockIdx.x * blockDim.x + threadIdx.x;
    if (i < n) p[i] = 0.f;
}

// --------------------------------- GEMM ------------------------------------
// Y[M,N] = A[M,K] @ W[N,K]^T, f16 operands, f32 accumulate.
// Block = 256 threads = 8 waves; block tile 128(M) x 128(N); wave tile 16x128.
// W tile (128 rows x 64 K-halves) double-buffered in LDS; fused column stats.

#define ROWH 72   // padded halves per W row slice in LDS (144 B, 16B-aligned)

__global__ __launch_bounds__(256) void gemm_lds_kernel(
    const half_t* __restrict__ A, const half_t* __restrict__ W,
    float* __restrict__ Y, float* __restrict__ gsum, float* __restrict__ gsumsq,
    int M, int N, int K) {
    __shared__ half_t lw[2][128 * ROWH];
    __shared__ float  lsum[128];
    __shared__ float  lsq[128];

    const int tid  = threadIdx.x;
    const int wave = tid >> 5;
    const int lane = tid & 31;
    const int ln16 = lane & 15;
    const int kg   = lane >> 4;

    const int mBase   = blockIdx.y * 128 + wave * 16;
    const int colBase = blockIdx.x * 128;

    if (tid < 128) { lsum[tid] = 0.f; lsq[tid] = 0.f; }

    const half_t* arow = A + (size_t)(mBase + ln16) * K;

    // cooperative staging: thread t copies 32 halves of W row (t>>1), half (t&1)
    const int srow = tid >> 1, ssel = tid & 1;
    const half_t* wsrc = W + (size_t)(colBase + srow) * K + ssel * 32;
    half_t* sdst[2] = { &lw[0][srow * ROWH + ssel * 32],
                        &lw[1][srow * ROWH + ssel * 32] };

    v8f acc[8] = {};

    // ---- prologue: stage K-halves [0,64) into LDS buf 0, load A-frags ----
    {
        v8h s0 = *(const v8h*)(wsrc + 0);
        v8h s1 = *(const v8h*)(wsrc + 8);
        v8h s2 = *(const v8h*)(wsrc + 16);
        v8h s3 = *(const v8h*)(wsrc + 24);
        *(v8h*)(sdst[0] + 0)  = s0;
        *(v8h*)(sdst[0] + 8)  = s1;
        *(v8h*)(sdst[0] + 16) = s2;
        *(v8h*)(sdst[0] + 24) = s3;
    }
    v16h aCur0 = cat16(*(const v8h*)(arow + 8 * kg),
                       *(const v8h*)(arow + 16 + 8 * kg));
    v16h aCur1 = cat16(*(const v8h*)(arow + 32 + 8 * kg),
                       *(const v8h*)(arow + 48 + 8 * kg));
    __syncthreads();

    const int nK2 = K >> 6;
    for (int ks = 0; ks < nK2; ++ks) {
        const half_t* curBuf = lw[ks & 1];
        half_t* nxtDst = sdst[(ks & 1) ^ 1];
        const bool more = (ks + 1 < nK2);
        const int kNext = (ks + 1) << 6;

        // issue next stage's global traffic early (overlaps both WMMA bursts)
        v8h s0, s1, s2, s3;
        v16h aNxt0, aNxt1;
        if (more) {
            __builtin_prefetch(arow + kNext + 192, 0, 3);
            s0 = *(const v8h*)(wsrc + kNext);
            s1 = *(const v8h*)(wsrc + kNext + 8);
            s2 = *(const v8h*)(wsrc + kNext + 16);
            s3 = *(const v8h*)(wsrc + kNext + 24);
            aNxt0 = cat16(*(const v8h*)(arow + kNext + 8 * kg),
                          *(const v8h*)(arow + kNext + 16 + 8 * kg));
            aNxt1 = cat16(*(const v8h*)(arow + kNext + 32 + 8 * kg),
                          *(const v8h*)(arow + kNext + 48 + 8 * kg));
        }

        // two K=32 sub-steps; per sub-step: bulk-preload 8 B-frags (all 16
        // ds_load_b128 issued before any WMMA -> graduated DScnt waits),
        // then an 8-WMMA burst.
#pragma unroll
        for (int s = 0; s < 2; ++s) {
            v16h bFrag[8];
#pragma unroll
            for (int t = 0; t < 8; ++t) {
                const half_t* bp = curBuf + (t * 16 + ln16) * ROWH + s * 32 + kg * 8;
                bFrag[t] = cat16(*(const v8h*)(bp), *(const v8h*)(bp + 16));
            }
            SCHED_FENCE();   // keep all DS reads above, all WMMAs below
            const v16h aS = s ? aCur1 : aCur0;
#pragma unroll
            for (int t = 0; t < 8; ++t)
                acc[t] = __builtin_amdgcn_wmma_f32_16x16x32_f16(
                    false, aS, false, bFrag[t], (short)0, acc[t], false, false);
#if __has_builtin(__builtin_amdgcn_sched_group_barrier)
            __builtin_amdgcn_sched_group_barrier(0x008, 8, 0);  // 8 matrix ops
#endif
        }

        if (more) {
            *(v8h*)(nxtDst + 0)  = s0;
            *(v8h*)(nxtDst + 8)  = s1;
            *(v8h*)(nxtDst + 16) = s2;
            *(v8h*)(nxtDst + 24) = s3;
            aCur0 = aNxt0;
            aCur1 = aNxt1;
        }
        __syncthreads();
    }

    // ---- epilogue: store Y, fold per-column sum/sumsq into LDS then global ----
#pragma unroll
    for (int t = 0; t < 8; ++t) {
        const int n = colBase + t * 16 + ln16;
        float s = 0.f, s2 = 0.f;
#pragma unroll
        for (int e = 0; e < 8; ++e) {
            float v = acc[t][e];
            Y[(size_t)(mBase + e + 8 * kg) * N + n] = v;
            s += v;
            s2 += v * v;
        }
        atomicAdd(&lsum[t * 16 + ln16], s);   // ds_add_f32
        atomicAdd(&lsq[t * 16 + ln16], s2);
    }
    __syncthreads();
    if (tid < 128) {
        atomicAdd(&gsum[colBase + tid], lsum[tid]);
        atomicAdd(&gsumsq[colBase + tid], lsq[tid]);
    }
}

// ------------------------- head: WMMA + fused sigmoid ----------------------
// N padded 10 -> 16; each wave: one 16x16 tile column, direct global loads.

__global__ __launch_bounds__(256) void head_kernel(
    const half_t* __restrict__ A, const half_t* __restrict__ W,
    float* __restrict__ OUT, int K) {
    const int tid  = threadIdx.x;
    const int wave = tid >> 5;
    const int lane = tid & 31;
    const int ln16 = lane & 15;
    const int kg   = lane >> 4;

    const int mBase = blockIdx.y * 128 + wave * 16;
    const half_t* arow = A + (size_t)(mBase + ln16) * K;
    const half_t* wrow = W + (size_t)ln16 * K + 8 * kg;

    v8f acc = {};
    for (int k = 0; k < K; k += 32) {
        __builtin_prefetch(arow + k + 128, 0, 3);
        v8h a0 = *(const v8h*)(arow + k + 8 * kg);
        v8h a1 = *(const v8h*)(arow + k + 16 + 8 * kg);
        v8h b0 = *(const v8h*)(wrow + k);
        v8h b1 = *(const v8h*)(wrow + k + 16);
        acc = __builtin_amdgcn_wmma_f32_16x16x32_f16(
            false, cat16(a0, a1), false, cat16(b0, b1), (short)0, acc, false, false);
    }
    if (ln16 < 10) {
#pragma unroll
        for (int e = 0; e < 8; ++e) {
            float v = acc[e];
            OUT[(size_t)(mBase + e + 8 * kg) * 10 + ln16] = 1.f / (1.f + __expf(-v));
        }
    }
}

// --------------------------- BatchNorm pipeline ----------------------------

__global__ void bnfin_kernel(const float* __restrict__ sum, const float* __restrict__ sumsq,
                             const float* __restrict__ g, const float* __restrict__ b,
                             float* __restrict__ scale, float* __restrict__ shift,
                             int N, float invM) {
    int j = blockIdx.x * blockDim.x + threadIdx.x;
    if (j < N) {
        float mean = sum[j] * invM;
        float var  = sumsq[j] * invM - mean * mean;   // biased variance (BN train mode)
        float sc   = g[j] * rsqrtf(var + BN_EPS);
        scale[j] = sc;
        shift[j] = b[j] - mean * sc;
    }
}

__global__ void bnrelu_kernel(const float* __restrict__ Y, const float* __restrict__ scale,
                              const float* __restrict__ shift, half_t* __restrict__ Ah,
                              size_t n, int N) {
    for (size_t i = (size_t)blockIdx.x * blockDim.x + threadIdx.x; i < n;
         i += (size_t)gridDim.x * blockDim.x) {
        int col = (int)(i & (size_t)(N - 1));         // N == 1024
        float v = fmaxf(0.f, Y[i] * scale[col] + shift[col]);
        Ah[i] = (half_t)v;
    }
}

// ------------------------------- launcher ----------------------------------

extern "C" void kernel_launch(void* const* d_in, const int* in_sizes, int n_in,
                              void* d_out, int out_size, void* d_ws, size_t ws_size,
                              hipStream_t stream) {
    const float* x  = (const float*)d_in[0];
    const float* w1 = (const float*)d_in[1];
    const float* g1 = (const float*)d_in[2];
    const float* b1 = (const float*)d_in[3];
    const float* w2 = (const float*)d_in[4];
    const float* g2 = (const float*)d_in[5];
    const float* b2 = (const float*)d_in[6];
    const float* w3 = (const float*)d_in[7];
    const float* g3 = (const float*)d_in[8];
    const float* b3 = (const float*)d_in[9];
    const float* w4 = (const float*)d_in[10];

    uint8_t* ws = (uint8_t*)d_ws;
    size_t off = 0;
    auto alloc = [&](size_t bytes) -> void* {
        void* p = ws + off;
        off = (off + bytes + 255) & ~(size_t)255;
        return p;
    };
    half_t* Xh    = (half_t*)alloc((size_t)BATCH * K1_PAD * sizeof(half_t));
    half_t* W1h   = (half_t*)alloc((size_t)HID * K1_PAD * sizeof(half_t));
    half_t* W2h   = (half_t*)alloc((size_t)HID * HID * sizeof(half_t));
    half_t* W3h   = (half_t*)alloc((size_t)HID * HID * sizeof(half_t));
    half_t* W4h   = (half_t*)alloc((size_t)16 * HID * sizeof(half_t));
    float*  Yb    = (float*)alloc((size_t)BATCH * HID * sizeof(float));
    half_t* Ah    = (half_t*)alloc((size_t)BATCH * HID * sizeof(half_t));
    float*  sum   = (float*)alloc(HID * sizeof(float));
    float*  sumsq = (float*)alloc(HID * sizeof(float));
    float*  scale = (float*)alloc(HID * sizeof(float));
    float*  shift = (float*)alloc(HID * sizeof(float));
    (void)ws_size; (void)in_sizes; (void)n_in; (void)out_size;

    const dim3 blk(256);
    const float invM = 1.f / (float)BATCH;

    // -------- prep: f16 cast of x (K-padded) + binarized f16 weights --------
    cast_pad_kernel<<<4096, blk, 0, stream>>>(x, Xh, BATCH, K1_IN, K1_PAD);
    binarize_pad_kernel<<<2048, blk, 0, stream>>>(w1, W1h, HID, HID, K1_IN, K1_PAD);
    binarize_pad_kernel<<<2048, blk, 0, stream>>>(w2, W2h, HID, HID, HID, HID);
    binarize_pad_kernel<<<2048, blk, 0, stream>>>(w3, W3h, HID, HID, HID, HID);
    binarize_pad_kernel<<<64,   blk, 0, stream>>>(w4, W4h, 10, 16, HID, HID);

    const dim3 gemmGrid(HID / 128, BATCH / 128);   // 8 x 128 blocks
    const size_t nAct = (size_t)BATCH * HID;

    // ------------------------------ layer 1 --------------------------------
    zero_kernel<<<4, blk, 0, stream>>>(sum, HID);
    zero_kernel<<<4, blk, 0, stream>>>(sumsq, HID);
    gemm_lds_kernel<<<gemmGrid, blk, 0, stream>>>(Xh, W1h, Yb, sum, sumsq,
                                                  BATCH, HID, K1_PAD);
    bnfin_kernel<<<4, blk, 0, stream>>>(sum, sumsq, g1, b1, scale, shift, HID, invM);
    bnrelu_kernel<<<8192, blk, 0, stream>>>(Yb, scale, shift, Ah, nAct, HID);

    // ------------------------------ layer 2 --------------------------------
    zero_kernel<<<4, blk, 0, stream>>>(sum, HID);
    zero_kernel<<<4, blk, 0, stream>>>(sumsq, HID);
    gemm_lds_kernel<<<gemmGrid, blk, 0, stream>>>(Ah, W2h, Yb, sum, sumsq,
                                                  BATCH, HID, HID);
    bnfin_kernel<<<4, blk, 0, stream>>>(sum, sumsq, g2, b2, scale, shift, HID, invM);
    bnrelu_kernel<<<8192, blk, 0, stream>>>(Yb, scale, shift, Ah, nAct, HID);

    // ------------------------------ layer 3 --------------------------------
    zero_kernel<<<4, blk, 0, stream>>>(sum, HID);
    zero_kernel<<<4, blk, 0, stream>>>(sumsq, HID);
    gemm_lds_kernel<<<gemmGrid, blk, 0, stream>>>(Ah, W3h, Yb, sum, sumsq,
                                                  BATCH, HID, HID);
    bnfin_kernel<<<4, blk, 0, stream>>>(sum, sumsq, g3, b3, scale, shift, HID, invM);
    bnrelu_kernel<<<8192, blk, 0, stream>>>(Yb, scale, shift, Ah, nAct, HID);

    // ----------------------- head: GEMM + fused sigmoid --------------------
    head_kernel<<<dim3(1, BATCH / 128), blk, 0, stream>>>(Ah, W4h, (float*)d_out, HID);
}
